// LinearAttention_35837207117907
// MI455X (gfx1250) — matrix operand order repaired and checked by reading, not compiled, not run on hardware
//
#include <hip/hip_runtime.h>
#include <hip/hip_bf16.h>

// Linear attention, MI455X (gfx1250, wave32).
// Heavy math: v_wmma_f32_16x16x32_bf16 (fp32 accum).
// Global->LDS staging: global_load_async_to_lds_b128 (ASYNCcnt path),
// double-buffered with peeled last iteration: steady-state loop is
// issue(i+1); s_wait_asynccnt 6; barrier; wmma; barrier  (branch-free).
// Shapes: B=4, N=4096, DIM=1024, H=16, DH=64, INNER=1024, BH=64.

typedef __bf16 bf16_t;
typedef __attribute__((ext_vector_type(16))) __bf16 v16bf;
typedef __attribute__((ext_vector_type(8)))  __bf16 v8bf;
typedef __attribute__((ext_vector_type(8)))  float  v8f;

#define LDT 40   // LDS tile pitch in halves (32 data + 8 pad) -> conflict-free 16B frag loads

static __device__ __forceinline__ float  bf2f(bf16_t x) { return (float)x; }
static __device__ __forceinline__ bf16_t f2bf(float x)  { return (bf16_t)x; }
static __device__ __forceinline__ float  elu1(float x)  { return x > 0.f ? x + 1.f : __expf(x); }

// ---- gfx1250 async global->LDS copy (ASYNCcnt) ----
static __device__ __forceinline__ unsigned lds_off(const void* p) {
  return (unsigned)(unsigned long long)p;  // generic addr low 32 bits = LDS byte offset
}
static __device__ __forceinline__ void async_b128(unsigned lds, const void* g) {
  asm volatile("global_load_async_to_lds_b128 %0, %1, off" :: "v"(lds), "v"(g) : "memory");
}
static __device__ __forceinline__ void async_wait_le6() {
  asm volatile("s_wait_asynccnt 0x6" ::: "memory");
}
static __device__ __forceinline__ void async_wait_le2() {
  asm volatile("s_wait_asynccnt 0x2" ::: "memory");
}
static __device__ __forceinline__ void async_wait_le0() {
  asm volatile("s_wait_asynccnt 0x0" ::: "memory");
}
// copy one 32B segment (16 halves) global->LDS (2 async instructions)
static __device__ __forceinline__ void async_seg(bf16_t* dp, const bf16_t* sp) {
  async_b128(lds_off(dp), sp);
  async_b128(lds_off(dp + 8), sp + 8);
}

// ---- WMMA fragment loaders (ISA 7.12.2 layouts), pitch in halves ----
static __device__ __forceinline__ v16bf load_fragA(const bf16_t* base, int pitch, int row0) {
  const int lane = threadIdx.x & 31;
  const int m = lane & 15, h = lane >> 4;
  const bf16_t* p = base + (row0 + m) * pitch + h * 8;
  v8bf lo = *(const v8bf*)p;
  v8bf hi = *(const v8bf*)(p + 16);
  v16bf r;
#pragma unroll
  for (int i = 0; i < 8; ++i) { r[i] = lo[i]; r[i + 8] = hi[i]; }
  return r;
}
static __device__ __forceinline__ v16bf load_fragB(const bf16_t* base, int pitch, int col0) {
  const int lane = threadIdx.x & 31;
  const int n = lane & 15, h = lane >> 4;
  const bf16_t* p = base + (col0 + n) * pitch + h * 8;
  v8bf lo = *(const v8bf*)p;
  v8bf hi = *(const v8bf*)(p + 16);
  v16bf r;
#pragma unroll
  for (int i = 0; i < 8; ++i) { r[i] = lo[i]; r[i + 8] = hi[i]; }
  return r;
}
static __device__ __forceinline__ v8f wmma_bf16(v16bf a, v16bf b, v8f c) {
  return __builtin_amdgcn_wmma_f32_16x16x32_bf16(false, a, false, b, (short)0, c, false, false);
}

// ---------------- K1a: f32 -> bf16 copy ----------------
__global__ __launch_bounds__(256) void cvt_kernel(const float* __restrict__ in,
                                                  bf16_t* __restrict__ out, int n) {
  for (int i = blockIdx.x * blockDim.x + threadIdx.x; i < n; i += gridDim.x * blockDim.x)
    out[i] = f2bf(in[i]);
}

// ---------------- K1b: f32 (K x N) -> bf16 transposed (N x K) ----------------
__global__ __launch_bounds__(256) void transpose_cvt_kernel(const float* __restrict__ in,
                                                            bf16_t* __restrict__ out,
                                                            int K, int N) {
  __shared__ float t[32][33];
  const int n0 = blockIdx.x * 32, k0 = blockIdx.y * 32;
  const int tx = threadIdx.x & 31, ty = threadIdx.x >> 5;
#pragma unroll
  for (int r = ty; r < 32; r += 8) t[r][tx] = in[(long)(k0 + r) * N + (n0 + tx)];
  __syncthreads();
#pragma unroll
  for (int r = ty; r < 32; r += 8) out[(long)(n0 + r) * K + (k0 + tx)] = f2bf(t[tx][r]);
}

// ---------------- K2: QKV GEMM + fused elu/mask epilogue ----------------
// C(16384x3072) = Xb * WqkvT^T ; block tile 128x256, wave tile 64x64; grid (12, 128).
__global__ __launch_bounds__(256) void qkv_gemm_kernel(
    const bf16_t* __restrict__ Xb, const bf16_t* __restrict__ WqkvT,
    const unsigned char* __restrict__ mask,
    bf16_t* __restrict__ qbuf,    // (BH, N, 64)
    bf16_t* __restrict__ kTbuf,   // (BH, 64, N)
    bf16_t* __restrict__ vTbuf) { // (BH, 64, N)
  __shared__ __align__(16) bf16_t sA[2][128 * LDT];
  __shared__ __align__(16) bf16_t sB[2][256 * LDT];
  const int row0 = blockIdx.y << 7;
  const int col0 = blockIdx.x << 8;
  const int wave = threadIdx.x >> 5;
  const int lane = threadIdx.x & 31;
  const int half = lane >> 4, ln = lane & 15;
  const int mBase = (wave >> 2) << 6;   // 0 or 64
  const int nBase = (wave & 3) << 6;    // 0,64,128,192
  const int tr = threadIdx.x >> 1, tc = (threadIdx.x & 1) << 4;

  v8f acc[4][4];
#pragma unroll
  for (int i = 0; i < 4; ++i)
#pragma unroll
    for (int j = 0; j < 4; ++j)
#pragma unroll
      for (int e = 0; e < 8; ++e) acc[i][j][e] = 0.f;

  auto issue = [&](int kb, int buf) {
    async_seg(sA[buf] + tr * LDT + tc, Xb + (long)(row0 + tr) * 1024 + kb + tc);
    async_seg(sB[buf] + tr * LDT + tc, WqkvT + (long)(col0 + tr) * 1024 + kb + tc);
    async_seg(sB[buf] + (128 + tr) * LDT + tc,
              WqkvT + (long)(col0 + 128 + tr) * 1024 + kb + tc);
  };
  auto compute = [&](int buf) {
    v16bf aF[4], bF[4];
#pragma unroll
    for (int mt = 0; mt < 4; ++mt) aF[mt] = load_fragA(sA[buf], LDT, mBase + (mt << 4));
#pragma unroll
    for (int nt = 0; nt < 4; ++nt) bF[nt] = load_fragB(sB[buf], LDT, nBase + (nt << 4));
#pragma unroll
    for (int mt = 0; mt < 4; ++mt)
#pragma unroll
      for (int nt = 0; nt < 4; ++nt) acc[mt][nt] = wmma_bf16(aF[mt], bF[nt], acc[mt][nt]);
  };

  issue(0, 0);
  for (int kb = 0; kb < 992; kb += 32) {   // steady state: branch-free wait_le6
    const int buf = (kb >> 5) & 1;
    issue(kb + 32, buf ^ 1);
    async_wait_le6();
    __syncthreads();
    compute(buf);
    __syncthreads();
  }
  async_wait_le0();                        // peeled last tile (kb=992, buf=1)
  __syncthreads();
  compute(1);

  const int sect = col0 >> 10;  // 0=q 1=k 2=v (uniform per block: 256 | 1024)
#pragma unroll
  for (int mt = 0; mt < 4; ++mt) {
#pragma unroll
    for (int j = 0; j < 8; ++j) {
      const int r = row0 + mBase + (mt << 4) + j + half * 8;
      const int b = r >> 12, n = r & 4095;
      const float mval = mask[(b << 12) + n] ? 1.f : 0.f;
#pragma unroll
      for (int nt = 0; nt < 4; ++nt) {
        const int c = col0 + nBase + (nt << 4) + ln;
        const int cc = c & 1023;
        const int h = cc >> 6, d = cc & 63;
        const long bh = (long)(b * 16 + h);
        float v = acc[mt][nt][j];
        if (sect == 0) {
          qbuf[(bh * 4096 + n) * 64 + d] = f2bf(elu1(v));
        } else if (sect == 1) {
          kTbuf[(bh * 64 + d) * 4096 + n] = f2bf(elu1(v) * mval);
        } else {
          vTbuf[(bh * 64 + d) * 4096 + n] = f2bf(v * mval);
        }
      }
    }
  }
}

// ---------------- K3: kv[bh] = K^T V  (64x64), written transposed kvT[e][d] ----------------
__global__ __launch_bounds__(256) void kv_gemm_kernel(const bf16_t* __restrict__ kT,
                                                      const bf16_t* __restrict__ vT,
                                                      bf16_t* __restrict__ kvT) {
  __shared__ __align__(16) bf16_t sA[2][64 * LDT];
  __shared__ __align__(16) bf16_t sB[2][64 * LDT];
  const int bh = blockIdx.x;
  const int wave = threadIdx.x >> 5;
  const int lane = threadIdx.x & 31;
  const int half = lane >> 4, ln = lane & 15;
  const int mBase = (wave & 3) << 4;   // d-tile
  const int nBase = (wave >> 2) << 5;  // e-tile pair
  const bf16_t* A  = kT + (long)bh * 64 * 4096;
  const bf16_t* Bm = vT + (long)bh * 64 * 4096;
  // staging map: 256 threads cover 2 tiles x 128 segments
  const int s = threadIdx.x & 127;
  const int sr = s >> 1, sc = (s & 1) << 4;
  const bf16_t* ssrc = (threadIdx.x < 128 ? A : Bm) + (long)sr * 4096 + sc;
  const int sdoff = sr * LDT + sc;

  v8f acc[2];
#pragma unroll
  for (int j = 0; j < 2; ++j)
#pragma unroll
    for (int e = 0; e < 8; ++e) acc[j][e] = 0.f;

  auto issue = [&](int kb, int buf) {
    bf16_t* d = (threadIdx.x < 128 ? sA[buf] : sB[buf]) + sdoff;
    async_seg(d, ssrc + kb);
  };
  auto compute = [&](int buf) {
    v16bf aF = load_fragA(sA[buf], LDT, mBase);
#pragma unroll
    for (int nt = 0; nt < 2; ++nt) {
      v16bf bF = load_fragB(sB[buf], LDT, nBase + (nt << 4));
      acc[nt] = wmma_bf16(aF, bF, acc[nt]);
    }
  };

  issue(0, 0);
  for (int kb = 0; kb < 4064; kb += 32) {
    const int buf = (kb >> 5) & 1;
    issue(kb + 32, buf ^ 1);
    async_wait_le2();
    __syncthreads();
    compute(buf);
    __syncthreads();
  }
  async_wait_le0();                        // peeled last tile (kb=4064, buf=1)
  __syncthreads();
  compute(1);

#pragma unroll
  for (int nt = 0; nt < 2; ++nt)
#pragma unroll
    for (int j = 0; j < 8; ++j) {
      const int d = mBase + j + half * 8;
      const int e = nBase + (nt << 4) + ln;
      kvT[(long)bh * 4096 + e * 64 + d] = f2bf(acc[nt][j]);
    }
}

// ---------------- K3b: ksum[bh][d] = sum_n kT[bh][d][n] ----------------
__global__ __launch_bounds__(64) void ksum_kernel(const bf16_t* __restrict__ kT,
                                                  float* __restrict__ ksum) {
  const int bh = blockIdx.x, d = threadIdx.x;
  const bf16_t* p = kT + ((long)bh * 64 + d) * 4096;
  float s = 0.f;
  for (int n = 0; n < 4096; ++n) s += bf2f(p[n]);
  ksum[bh * 64 + d] = s;
}

// ---------------- K4: out = (Q * kv) * z, fused normalize; grid (32, 64) ----------------
__global__ __launch_bounds__(256) void out_gemm_kernel(const bf16_t* __restrict__ qbuf,
                                                       const bf16_t* __restrict__ kvT,
                                                       const float* __restrict__ ksum,
                                                       bf16_t* __restrict__ attn) { // (B,N,1024)
  __shared__ float sz[128];
  const int bh = blockIdx.y;
  const int b = bh >> 4, h = bh & 15;
  const int n0 = blockIdx.x << 7;
  const int wave = threadIdx.x >> 5;
  const int lane = threadIdx.x & 31;
  const int half = lane >> 4, ln = lane & 15;

  if (threadIdx.x < 128) {
    const bf16_t* qr = qbuf + ((long)bh * 4096 + n0 + threadIdx.x) * 64;
    const float* ks = ksum + bh * 64;
    float s = 0.f;
#pragma unroll
    for (int d = 0; d < 64; ++d) s += bf2f(qr[d]) * ks[d];
    sz[threadIdx.x] = 1.f / (s + 1e-6f);
  }
  __syncthreads();

  v8f acc[4];
#pragma unroll
  for (int j = 0; j < 4; ++j)
#pragma unroll
    for (int e = 0; e < 8; ++e) acc[j][e] = 0.f;

#pragma unroll
  for (int kb = 0; kb < 64; kb += 32) {
    v16bf aF = load_fragA(qbuf + ((long)bh * 4096 + n0) * 64 + kb, 64, wave << 4);
#pragma unroll
    for (int et = 0; et < 4; ++et) {
      v16bf bF = load_fragB(kvT + (long)bh * 4096 + kb, 64, et << 4);
      acc[et] = wmma_bf16(aF, bF, acc[et]);
    }
  }
#pragma unroll
  for (int et = 0; et < 4; ++et)
#pragma unroll
    for (int j = 0; j < 8; ++j) {
      const int lrow = (wave << 4) + j + half * 8;
      const int n = n0 + lrow;
      const int e = (et << 4) + ln;
      const float v = acc[et][j] * sz[lrow];
      attn[((long)(b * 4096 + n)) * 1024 + h * 64 + e] = f2bf(v);
    }
}

// ---------------- K5: final GEMM + bias, fp32 out; block 128x256; grid (4, 128) ----------------
__global__ __launch_bounds__(256) void final_gemm_kernel(const bf16_t* __restrict__ attn,
                                                         const bf16_t* __restrict__ WoutT,
                                                         const float* __restrict__ bout,
                                                         float* __restrict__ out) {
  __shared__ __align__(16) bf16_t sA[2][128 * LDT];
  __shared__ __align__(16) bf16_t sB[2][256 * LDT];
  const int row0 = blockIdx.y << 7;
  const int col0 = blockIdx.x << 8;
  const int wave = threadIdx.x >> 5;
  const int lane = threadIdx.x & 31;
  const int half = lane >> 4, ln = lane & 15;
  const int mBase = (wave >> 2) << 6;
  const int nBase = (wave & 3) << 6;
  const int tr = threadIdx.x >> 1, tc = (threadIdx.x & 1) << 4;

  v8f acc[4][4];
#pragma unroll
  for (int i = 0; i < 4; ++i)
#pragma unroll
    for (int j = 0; j < 4; ++j)
#pragma unroll
      for (int e = 0; e < 8; ++e) acc[i][j][e] = 0.f;

  auto issue = [&](int kb, int buf) {
    async_seg(sA[buf] + tr * LDT + tc, attn + (long)(row0 + tr) * 1024 + kb + tc);
    async_seg(sB[buf] + tr * LDT + tc, WoutT + (long)(col0 + tr) * 1024 + kb + tc);
    async_seg(sB[buf] + (128 + tr) * LDT + tc,
              WoutT + (long)(col0 + 128 + tr) * 1024 + kb + tc);
  };
  auto compute = [&](int buf) {
    v16bf aF[4], bF[4];
#pragma unroll
    for (int mt = 0; mt < 4; ++mt) aF[mt] = load_fragA(sA[buf], LDT, mBase + (mt << 4));
#pragma unroll
    for (int nt = 0; nt < 4; ++nt) bF[nt] = load_fragB(sB[buf], LDT, nBase + (nt << 4));
#pragma unroll
    for (int mt = 0; mt < 4; ++mt)
#pragma unroll
      for (int nt = 0; nt < 4; ++nt) acc[mt][nt] = wmma_bf16(aF[mt], bF[nt], acc[mt][nt]);
  };

  issue(0, 0);
  for (int kb = 0; kb < 992; kb += 32) {
    const int buf = (kb >> 5) & 1;
    issue(kb + 32, buf ^ 1);
    async_wait_le6();
    __syncthreads();
    compute(buf);
    __syncthreads();
  }
  async_wait_le0();
  __syncthreads();
  compute(1);

#pragma unroll
  for (int mt = 0; mt < 4; ++mt)
#pragma unroll
    for (int nt = 0; nt < 4; ++nt)
#pragma unroll
      for (int j = 0; j < 8; ++j) {
        const int r = row0 + mBase + (mt << 4) + j + half * 8;
        const int c = col0 + nBase + (nt << 4) + ln;
        out[(long)r * 1024 + c] = acc[mt][nt][j] + bout[c];
      }
}

// ---------------- launch ----------------
extern "C" void kernel_launch(void* const* d_in, const int* in_sizes, int n_in,
                              void* d_out, int out_size, void* d_ws, size_t ws_size,
                              hipStream_t stream) {
  const float* x            = (const float*)d_in[0];
  const unsigned char* mask = (const unsigned char*)d_in[1];  // jax bool = 1 byte
  const float* Wqkv         = (const float*)d_in[2];
  const float* Wout         = (const float*)d_in[3];
  const float* bout         = (const float*)d_in[4];
  float* out                = (float*)d_out;

  char* w = (char*)d_ws;  // ~177 MB used
  bf16_t* xb    = (bf16_t*)(w);                    // 16384*1024
  bf16_t* wqkvT = (bf16_t*)(w + 33554432ll);       // 3072*1024
  bf16_t* woutT = (bf16_t*)(w + 39845888ll);       // 1024*1024
  bf16_t* qbuf  = (bf16_t*)(w + 41943040ll);       // 64*4096*64
  bf16_t* kTbuf = (bf16_t*)(w + 75497472ll);       // 64*64*4096
  bf16_t* vTbuf = (bf16_t*)(w + 109051904ll);      // 64*64*4096
  bf16_t* kvT   = (bf16_t*)(w + 142606336ll);      // 64*64*64
  float*  ksum  = (float*) (w + 143130624ll);      // 64*64
  bf16_t* attn  = (bf16_t*)(w + 143147008ll);      // 16384*1024

  cvt_kernel<<<4096, 256, 0, stream>>>(x, xb, 16384 * 1024);
  transpose_cvt_kernel<<<dim3(96, 32), 256, 0, stream>>>(Wqkv, wqkvT, 1024, 3072);
  transpose_cvt_kernel<<<dim3(32, 32), 256, 0, stream>>>(Wout, woutT, 1024, 1024);
  qkv_gemm_kernel<<<dim3(12, 128), 256, 0, stream>>>(xb, wqkvT, mask, qbuf, kTbuf, vTbuf);
  kv_gemm_kernel<<<64, 256, 0, stream>>>(kTbuf, vTbuf, kvT);
  ksum_kernel<<<64, 64, 0, stream>>>(kTbuf, ksum);
  out_gemm_kernel<<<dim3(32, 64), 256, 0, stream>>>(qbuf, kvT, ksum, attn);
  final_gemm_kernel<<<dim3(4, 128), 256, 0, stream>>>(attn, woutT, bout, out);
}
